// SP_CAM_Model2_16140487098910
// MI455X (gfx1250) — matrix operand order repaired and checked by reading, not compiled, not run on hardware
//
#include <hip/hip_runtime.h>
#include <hip/hip_bf16.h>

// =====================================================================
// SP_CAM model on MI455X (gfx1250): all conv/matmul work lowered to
// V_WMMA_F32_16X16X32_BF16 implicit-GEMM (bf16 inputs, fp32 accumulate),
// fused BN/bias/residual/ReLU epilogues. Weight tiles are staged into
// LDS with GLOBAL_LOAD_ASYNC_TO_LDS_B128 (CDNA5 async path) when the
// toolchain exposes the builtin, else vectorized B128 sync copies.
// Wave32, 4 waves/block, 64x64 block tile, 32x32 wave tile, K-step 32.
// =====================================================================

typedef __attribute__((ext_vector_type(16))) __bf16 v16bf;
typedef __attribute__((ext_vector_type(8)))  float  v8f;
typedef __attribute__((ext_vector_type(4)))  int    v4i_;

#if defined(__HIP_DEVICE_COMPILE__) && __has_builtin(__builtin_amdgcn_global_load_async_to_lds_b128)
#define ASYNC_LDS 1
#else
#define ASYNC_LDS 0
#endif

#if ASYNC_LDS
__device__ __forceinline__ void async_copy16(const void* g, void* l) {
  // param0: int4 AS(1)* (global src), param1: int4 AS(3)* (LDS dst),
  // then imm offset, imm cpol  (types per hipcc diagnostic, round 2)
  __builtin_amdgcn_global_load_async_to_lds_b128(
      (__attribute__((address_space(1))) v4i_*)g,
      (__attribute__((address_space(3))) v4i_*)l, 0, 0);
}
__device__ __forceinline__ void async_wait0() {
#if __has_builtin(__builtin_amdgcn_s_wait_asynccnt)
  __builtin_amdgcn_s_wait_asynccnt(0);
#else
  asm volatile("s_wait_asynccnt 0x0" ::: "memory");
#endif
}
#endif

__device__ __forceinline__ unsigned short f2bf(float f) {
  unsigned int u = __float_as_uint(f);
  unsigned int r = u + 0x7FFFu + ((u >> 16) & 1u);   // round-to-nearest-even
  return (unsigned short)(r >> 16);
}

// ---------------------------------------------------------------------
// Packing / elementwise kernels
// ---------------------------------------------------------------------
__global__ void pack_bf16_kernel(const float* __restrict__ s,
                                 unsigned short* __restrict__ d, long n) {
  long i = (long)blockIdx.x * blockDim.x + threadIdx.x;
  if (i < n) d[i] = f2bf(s[i]);
}

__global__ void concat2_pack_kernel(const float* __restrict__ a,
                                    const float* __restrict__ b,
                                    unsigned short* __restrict__ d,
                                    int Bn, int C1, int C2, long HW) {
  long i = (long)blockIdx.x * blockDim.x + threadIdx.x;
  long tot = (long)Bn * (C1 + C2) * HW;
  if (i >= tot) return;
  long s = i % HW;
  long t = i / HW;
  int ch = (int)(t % (C1 + C2));
  int bb = (int)(t / (C1 + C2));
  float v = (ch < C1) ? a[((long)bb * C1 + ch) * HW + s]
                      : b[((long)bb * C2 + (ch - C1)) * HW + s];
  d[i] = f2bf(v);
}

__global__ void maxpool_kernel(const float* __restrict__ x, float* __restrict__ y,
                               int BC, int H, int W, int OH, int OW) {
  long i = (long)blockIdx.x * blockDim.x + threadIdx.x;
  long tot = (long)BC * OH * OW;
  if (i >= tot) return;
  int ox = (int)(i % OW);
  long t = i / OW;
  int oy = (int)(t % OH);
  int bc = (int)(t / OH);
  const float* xb = x + (long)bc * H * W;
  float m = -3.402823466e38f;
  for (int ky = 0; ky < 3; ++ky) {
    int iy = oy * 2 - 1 + ky;
    if (iy < 0 || iy >= H) continue;
    for (int kx = 0; kx < 3; ++kx) {
      int ix = ox * 2 - 1 + kx;
      if (ix < 0 || ix >= W) continue;
      m = fmaxf(m, xb[(long)iy * W + ix]);
    }
  }
  y[i] = m;
}

__global__ void softmax9_kernel(const float* __restrict__ t, float* __restrict__ o,
                                int Bn, long HW) {
  long i = (long)blockIdx.x * blockDim.x + threadIdx.x;
  if (i >= (long)Bn * HW) return;
  int b = (int)(i / HW);
  long s = i % HW;
  const float* ti = t + (long)b * 9 * HW + s;
  float m = ti[0];
  for (int k = 1; k < 9; ++k) m = fmaxf(m, ti[(long)k * HW]);
  float e[9], sum = 0.f;
  for (int k = 0; k < 9; ++k) { e[k] = __expf(ti[(long)k * HW] - m); sum += e[k]; }
  float inv = 1.f / sum;
  float* ob = o + (long)b * 9 * HW + s;
  for (int k = 0; k < 9; ++k) ob[(long)k * HW] = e[k] * inv;
}

__global__ void upfeat_kernel(const float* __restrict__ x, const float* __restrict__ pr,
                              float* __restrict__ y, int Bn, int C, int H, int W) {
  long i = (long)blockIdx.x * blockDim.x + threadIdx.x;
  long tot = (long)Bn * C * H * W;
  if (i >= tot) return;
  int xw = (int)(i % W);
  long t = i / W;
  int yh = (int)(t % H); t /= H;
  int cc = (int)(t % C);
  int b  = (int)(t / C);
  const long HW = (long)H * W;
  const float* xb = x + ((long)b * C + cc) * HW;
  const float* pb = pr + (long)b * 9 * HW + (long)yh * W + xw;
  float acc = 0.f;
  int k = 0;
  for (int dy = 0; dy < 3; ++dy)
    for (int dx = 0; dx < 3; ++dx, ++k) {
      int yy = yh + dy - 1, xx = xw + dx - 1;
      if (yy >= 0 && yy < H && xx >= 0 && xx < W)
        acc += xb[(long)yy * W + xx] * pb[(long)k * HW];
    }
  y[i] = acc;
}

__global__ void avgpool_kernel(const float* __restrict__ x, float* __restrict__ y,
                               int BC, long HW) {
  int i = blockIdx.x * blockDim.x + threadIdx.x;
  if (i >= BC) return;
  const float* xb = x + (long)i * HW;
  float s = 0.f;
  for (long k = 0; k < HW; ++k) s += xb[k];
  y[i] = s / (float)HW;
}

// column-wise (over channels) L2-normalize + pack to bf16
__global__ void l2norm_pack_kernel(const float* __restrict__ x,
                                   unsigned short* __restrict__ y,
                                   int Bn, int C, long HW) {
  long i = (long)blockIdx.x * blockDim.x + threadIdx.x;
  if (i >= (long)Bn * HW) return;
  int b = (int)(i / HW);
  long s = i % HW;
  const float* xb = x + (long)b * C * HW + s;
  float ss = 0.f;
  for (int c = 0; c < C; ++c) { float v = xb[(long)c * HW]; ss += v * v; }
  float inv = 1.f / fmaxf(sqrtf(ss), 1e-12f);
  unsigned short* yb = y + (long)b * C * HW + s;
  for (int c = 0; c < C; ++c) yb[(long)c * HW] = f2bf(xb[(long)c * HW] * inv);
}

// aff[i,j] /= sum_i aff[i,j]   (one thread per column j)
__global__ void colnorm_kernel(float* __restrict__ aff, int Bn, int N) {
  long j = (long)blockIdx.x * blockDim.x + threadIdx.x;
  if (j >= (long)Bn * N) return;
  int b = (int)(j / N);
  int col = (int)(j % N);
  float* a = aff + (long)b * N * N + col;
  float s = 0.f;
  for (int i = 0; i < N; ++i) s += a[(long)i * N];
  float inv = 1.f / s;
  for (int i = 0; i < N; ++i) a[(long)i * N] *= inv;
}

// ---------------------------------------------------------------------
// WMMA implicit-GEMM convolution (NCHW, fused bias/BN/residual/ReLU)
// ---------------------------------------------------------------------
struct ConvP {
  const unsigned short* W;   // bf16 [Cout][Cin*KH*KW]  (row-major M x K)
  const unsigned short* X;   // bf16 [B][Cin][H][W]
  float* Y;                  // f32  [B][Cout][OH][OW]
  const float* cb;           // conv bias or null
  const float* sc;           // BN scale or null
  const float* bi;           // BN bias (valid iff sc)
  const float* res;          // residual (same layout as Y) or null
  int B, Cin, H, Wd, Cout, KH, KW, str, pad, OH, OW, relu;
};

union FragBF { v16bf v; unsigned int d[8]; };

__global__ __launch_bounds__(128) void conv_wmma_kernel(ConvP p) {
  // rows are 80B apart -> every 16B chunk is B128-aligned for async copies
  __shared__ unsigned short lA[64][40];   // [M][K] K-contiguous
  __shared__ unsigned short lB[64][40];   // [N][K] K-contiguous
  const int tid  = threadIdx.x;
  const int wave = tid >> 5, lane = tid & 31;
  const int half = lane >> 4, lrow = lane & 15;
  const int waveM = (wave & 1) * 32, waveN = (wave >> 1) * 32;
  const int m0 = blockIdx.y * 64;
  const long n0 = (long)blockIdx.x * 64;
  const int KHW = p.KH * p.KW;
  const int Ktot = p.Cin * KHW;
  const long Ntot = (long)p.B * p.OH * p.OW;
  const bool fastA = ((Ktot & 7) == 0) && (m0 + 64 <= p.Cout);

  v8f acc[2][2] = {};

  for (int k0 = 0; k0 < Ktot; k0 += 32) {
    const bool fullK = (k0 + 32 <= Ktot);
    // --- stage A (weights): contiguous M x K rows ---
    if (fastA && fullK) {
      // 64 rows x 64B = 256 x B128 chunks; 2 per thread
      for (int i = tid; i < 256; i += 128) {
        int r = i >> 2, ch = i & 3;
        const unsigned short* gsrc = p.W + (long)(m0 + r) * Ktot + k0 + ch * 8;
#if ASYNC_LDS
        async_copy16(gsrc, &lA[r][ch * 8]);
#else
        *(uint4*)&lA[r][ch * 8] = *(const uint4*)gsrc;
#endif
      }
    } else {
      for (int i = tid; i < 64 * 32; i += 128) {
        int r = i >> 5, c = i & 31;
        int m = m0 + r, k = k0 + c;
        unsigned short v = 0;
        if (m < p.Cout && k < Ktot) v = p.W[(long)m * Ktot + k];
        lA[r][c] = v;
      }
    }
    // prefetch next weight K-slice into cache hierarchy
    {
      int pm = m0 + (tid >> 1);
      if (k0 + 32 < Ktot && pm < p.Cout)
        __builtin_prefetch(&p.W[(long)pm * Ktot + k0 + 32], 0, 0);
    }
    // --- stage B (implicit im2col gather with zero padding) ---
    for (int i = tid; i < 64 * 32; i += 128) {
      int r = i >> 5, c = i & 31;
      long n = n0 + r;
      int k = k0 + c;
      unsigned short v = 0;
      if (n < Ntot && k < Ktot) {
        long t = n;
        int ox = (int)(t % p.OW); t /= p.OW;
        int oy = (int)(t % p.OH);
        int b  = (int)(t / p.OH);
        int ci, iy, ix;
        if (KHW == 1) {               // 1x1 conv: no kh/kw decode
          ci = k;
          iy = oy * p.str - p.pad;
          ix = ox * p.str - p.pad;
        } else {
          ci = k / KHW;
          int rem = k - ci * KHW;
          int kh = rem / p.KW;
          int kw = rem - kh * p.KW;
          iy = oy * p.str - p.pad + kh;
          ix = ox * p.str - p.pad + kw;
        }
        if (iy >= 0 && iy < p.H && ix >= 0 && ix < p.Wd)
          v = p.X[(((long)b * p.Cin + ci) * p.H + iy) * p.Wd + ix];
      }
      lB[r][c] = v;
    }
#if ASYNC_LDS
    if (fastA && fullK) async_wait0();
#endif
    __syncthreads();

    // --- fragment assembly per ISA VGPR layouts, then 2x2 WMMA ---
    FragBF fa[2], fb[2];
#pragma unroll
    for (int mi = 0; mi < 2; ++mi) {
      int row = waveM + mi * 16 + lrow;
      const unsigned int* src = (const unsigned int*)&lA[row][0];
#pragma unroll
      for (int vv = 0; vv < 8; ++vv) {
        int kk = (vv < 4 ? 2 * vv : 16 + 2 * (vv - 4)) + 8 * half;
        fa[mi].d[vv] = src[kk >> 1];
      }
    }
#pragma unroll
    for (int ni = 0; ni < 2; ++ni) {
      int row = waveN + ni * 16 + lrow;
      const unsigned int* src = (const unsigned int*)&lB[row][0];
#pragma unroll
      for (int vv = 0; vv < 8; ++vv) {
        int kk = (vv < 4 ? 2 * vv : 16 + 2 * (vv - 4)) + 8 * half;
        fb[ni].d[vv] = src[kk >> 1];
      }
    }
#pragma unroll
    for (int mi = 0; mi < 2; ++mi)
#pragma unroll
      for (int ni = 0; ni < 2; ++ni)
        acc[mi][ni] = __builtin_amdgcn_wmma_f32_16x16x32_bf16(
            false, fa[mi].v, false, fb[ni].v, (short)0, acc[mi][ni], false, false);
    __syncthreads();
  }

  // --- fused epilogue: conv-bias -> BN affine -> residual -> ReLU ---
#pragma unroll
  for (int mi = 0; mi < 2; ++mi) {
    int mbase = m0 + waveM + mi * 16 + 8 * half;
#pragma unroll
    for (int ni = 0; ni < 2; ++ni) {
      long n = n0 + waveN + ni * 16 + lrow;
      if (n >= Ntot) continue;
      long t = n;
      int ox = (int)(t % p.OW); t /= p.OW;
      int oy = (int)(t % p.OH);
      int b  = (int)(t / p.OH);
#pragma unroll
      for (int r = 0; r < 8; ++r) {
        int cm = mbase + r;
        if (cm >= p.Cout) continue;
        float v = acc[mi][ni][r];
        if (p.cb) v += p.cb[cm];
        if (p.sc) v = v * p.sc[cm] + p.bi[cm];
        long oi = (((long)b * p.Cout + cm) * p.OH + oy) * p.OW + ox;
        if (p.res) v += p.res[oi];
        if (p.relu && v < 0.f) v = 0.f;
        p.Y[oi] = v;
      }
    }
  }
}

// ---------------------------------------------------------------------
// WMMA strided batched GEMM  C[b] = op(A[b] * B[b])  (bf16 in, f32 out)
// A elem (m,k) at A[m*rsA + k*csA]; B elem (k,n) at B[k*rsB + n*csB].
// op: 0 = none, 1 = elementwise square (for aff = (x^T x)^2).
// ---------------------------------------------------------------------
struct GemmP {
  const unsigned short* A;
  const unsigned short* B;
  float* C;
  long rsA, csA, rsB, csB;
  long bA, bB, bC;
  int M, N, K, op;
};

__global__ __launch_bounds__(128) void gemm_wmma_kernel(GemmP g) {
  __shared__ unsigned short lA[64][40];
  __shared__ unsigned short lB[64][40];
  const unsigned short* Ab = g.A + (long)blockIdx.z * g.bA;
  const unsigned short* Bb = g.B + (long)blockIdx.z * g.bB;
  float* Cb = g.C + (long)blockIdx.z * g.bC;
  const int tid  = threadIdx.x;
  const int wave = tid >> 5, lane = tid & 31;
  const int half = lane >> 4, lrow = lane & 15;
  const int waveM = (wave & 1) * 32, waveN = (wave >> 1) * 32;
  const int m0 = blockIdx.y * 64;
  const int n0 = blockIdx.x * 64;
  const bool fastA = (g.csA == 1) && ((g.rsA & 7) == 0) && (m0 + 64 <= g.M);

  v8f acc[2][2] = {};

  for (int k0 = 0; k0 < g.K; k0 += 32) {
    const bool fullK = (k0 + 32 <= g.K);
    if (fastA && fullK) {
      for (int i = tid; i < 256; i += 128) {
        int r = i >> 2, ch = i & 3;
        const unsigned short* gsrc = Ab + (long)(m0 + r) * g.rsA + k0 + ch * 8;
#if ASYNC_LDS
        async_copy16(gsrc, &lA[r][ch * 8]);
#else
        *(uint4*)&lA[r][ch * 8] = *(const uint4*)gsrc;
#endif
      }
    } else {
      for (int i = tid; i < 64 * 32; i += 128) {
        int r = i >> 5, c = i & 31;
        int m = m0 + r, k = k0 + c;
        lA[r][c] = (m < g.M && k < g.K)
                       ? Ab[(long)m * g.rsA + (long)k * g.csA] : (unsigned short)0;
      }
    }
    for (int i = tid; i < 64 * 32; i += 128) {
      int r = i >> 5, c = i & 31;
      int n = n0 + r, k = k0 + c;
      lB[r][c] = (n < g.N && k < g.K)
                     ? Bb[(long)k * g.rsB + (long)n * g.csB] : (unsigned short)0;
    }
#if ASYNC_LDS
    if (fastA && fullK) async_wait0();
#endif
    __syncthreads();
    FragBF fa[2], fb[2];
#pragma unroll
    for (int mi = 0; mi < 2; ++mi) {
      int row = waveM + mi * 16 + lrow;
      const unsigned int* src = (const unsigned int*)&lA[row][0];
#pragma unroll
      for (int vv = 0; vv < 8; ++vv) {
        int kk = (vv < 4 ? 2 * vv : 16 + 2 * (vv - 4)) + 8 * half;
        fa[mi].d[vv] = src[kk >> 1];
      }
    }
#pragma unroll
    for (int ni = 0; ni < 2; ++ni) {
      int row = waveN + ni * 16 + lrow;
      const unsigned int* src = (const unsigned int*)&lB[row][0];
#pragma unroll
      for (int vv = 0; vv < 8; ++vv) {
        int kk = (vv < 4 ? 2 * vv : 16 + 2 * (vv - 4)) + 8 * half;
        fb[ni].d[vv] = src[kk >> 1];
      }
    }
#pragma unroll
    for (int mi = 0; mi < 2; ++mi)
#pragma unroll
      for (int ni = 0; ni < 2; ++ni)
        acc[mi][ni] = __builtin_amdgcn_wmma_f32_16x16x32_bf16(
            false, fa[mi].v, false, fb[ni].v, (short)0, acc[mi][ni], false, false);
    __syncthreads();
  }

#pragma unroll
  for (int mi = 0; mi < 2; ++mi) {
    int mbase = m0 + waveM + mi * 16 + 8 * half;
#pragma unroll
    for (int ni = 0; ni < 2; ++ni) {
      int n = n0 + waveN + ni * 16 + lrow;
      if (n >= g.N) continue;
#pragma unroll
      for (int r = 0; r < 8; ++r) {
        int m = mbase + r;
        if (m >= g.M) continue;
        float v = acc[mi][ni][r];
        if (g.op == 1) v = v * v;
        Cb[(long)m * g.N + n] = v;
      }
    }
  }
}

// =====================================================================
// Host orchestration
// =====================================================================
struct BlockP {
  const float *w1, *w2, *w3, *s1, *b1, *s2, *b2, *s3, *b3, *wd, *sd, *bd;
  bool down;
};
struct HC { const float *w, *b, *s, *bb; };

struct Ctx {
  hipStream_t st;
  unsigned short* xb;   // packed activation scratch (bf16)
  unsigned short* wb;   // packed weight scratch (bf16)
};

static inline int outDim(int H, int k, int s, int p) { return (H + 2 * p - k) / s + 1; }

static void launch_pack(const float* s, unsigned short* d, long n, hipStream_t st) {
  pack_bf16_kernel<<<(unsigned)((n + 255) / 256), 256, 0, st>>>(s, d, n);
}

// conv with X already packed into c.xb
static void convRaw(Ctx& c, const float* W, const float* cb, const float* sc,
                    const float* bi, const float* res, float* Y,
                    int B, int Cin, int H, int Wd, int Cout, int K, int str,
                    int pad, int relu) {
  long nw = (long)Cout * Cin * K * K;
  launch_pack(W, c.wb, nw, c.st);
  int OH = outDim(H, K, str, pad), OW = outDim(Wd, K, str, pad);
  ConvP p{c.wb, c.xb, Y, cb, sc, bi, res,
          B, Cin, H, Wd, Cout, K, K, str, pad, OH, OW, relu};
  long Ntot = (long)B * OH * OW;
  dim3 g((unsigned)((Ntot + 63) / 64), (unsigned)((Cout + 63) / 64), 1);
  conv_wmma_kernel<<<g, 128, 0, c.st>>>(p);
}

static void convOp(Ctx& c, const float* X, const float* W, const float* cb,
                   const float* sc, const float* bi, const float* res, float* Y,
                   int B, int Cin, int H, int Wd, int Cout, int K, int str,
                   int pad, int relu) {
  launch_pack(X, c.xb, (long)B * Cin * H * Wd, c.st);
  convRaw(c, W, cb, sc, bi, res, Y, B, Cin, H, Wd, Cout, K, str, pad, relu);
}

static void bneck(Ctx& c, const BlockP& bp, const float* x, float* out,
                  float* tmp, float* scb, int B, int Cin, int H, int W,
                  int mid, int stride) {
  // conv1 1x1 + BN + ReLU
  convOp(c, x, bp.w1, nullptr, bp.s1, bp.b1, nullptr, tmp,
         B, Cin, H, W, mid, 1, 1, 0, 1);
  // conv2 3x3 (stride) + BN + ReLU  (reads packed copy; in-place f32 OK)
  int OH = outDim(H, 3, stride, 1), OW = outDim(W, 3, stride, 1);
  convOp(c, tmp, bp.w2, nullptr, bp.s2, bp.b2, nullptr, tmp,
         B, mid, H, W, mid, 3, stride, 1, 1);
  // shortcut
  const float* res;
  if (bp.down) {
    convOp(c, x, bp.wd, nullptr, bp.sd, bp.bd, nullptr, scb,
           B, Cin, H, W, mid * 4, 1, stride, 0, 0);
    res = scb;
  } else {
    res = x;
  }
  // conv3 1x1 + BN + residual + ReLU
  convOp(c, tmp, bp.w3, nullptr, bp.s3, bp.b3, res, out,
         B, mid, OH, OW, mid * 4, 1, 1, 0, 1);
}

extern "C" void kernel_launch(void* const* d_in, const int* in_sizes, int n_in,
                              void* d_out, int out_size, void* d_ws, size_t ws_size,
                              hipStream_t stream) {
  (void)in_sizes; (void)n_in; (void)out_size; (void)ws_size;
  // ---- parameter cursor (dict flattening order) ----
  int ci = 0;
  auto nx = [&]() { return (const float*)d_in[ci++]; };
  const float* inputs = nx();
  const float* probs  = nx();
  const float* c1w = nx();
  const float* s0 = nx();
  const float* b0 = nx();
  auto readBlock = [&](bool down) {
    BlockP p{};
    p.w1 = nx(); p.w2 = nx(); p.w3 = nx();
    p.s1 = nx(); p.b1 = nx(); p.s2 = nx(); p.b2 = nx(); p.s3 = nx(); p.b3 = nx();
    p.down = down;
    if (down) { p.wd = nx(); p.sd = nx(); p.bd = nx(); }
    return p;
  };
  BlockP L1[3], L2[4], L3[6], L4[3];
  for (int i = 0; i < 3; ++i) L1[i] = readBlock(i == 0);
  for (int i = 0; i < 4; ++i) L2[i] = readBlock(i == 0);
  for (int i = 0; i < 6; ++i) L3[i] = readBlock(i == 0);
  for (int i = 0; i < 3; ++i) L4[i] = readBlock(i == 0);
  HC qf[3];
  for (int i = 0; i < 3; ++i) { qf[i].w = nx(); qf[i].b = nx(); qf[i].s = nx(); qf[i].bb = nx(); }
  HC x4f{nx(), nx(), nx(), nx()};
  HC x5f{nx(), nx(), nx(), nx()};
  HC tc4[2] = {{nx(), nx(), nullptr, nullptr}, {nx(), nx(), nullptr, nullptr}};
  HC tc5[2] = {{nx(), nx(), nullptr, nullptr}, {nx(), nx(), nullptr, nullptr}};
  const float* clsw = nx();
  // pcm / it scalars follow; hardcoded (pcm=2 -> square, it=2) per setup.

  // ---- workspace carve ----
  char* wsb = (char*)d_ws;
  size_t off = 0;
  auto carve = [&](size_t bytes) -> void* {
    void* p = wsb + off;
    off += (bytes + 255) & ~(size_t)255;
    return p;
  };
  const size_t MB = 1ull << 20;
  float* P0 = (float*)carve(34 * MB);
  float* P1 = (float*)carve(34 * MB);
  float* T  = (float*)carve(18 * MB);
  float* S  = (float*)carve(34 * MB);
  unsigned short* XB = (unsigned short*)carve(18 * MB);
  unsigned short* WB = (unsigned short*)carve(6 * MB);
  float* Q   = (float*)carve(2ull * 128 * 1024 * 4);
  float* F   = (float*)carve(2ull * 128 * 1024 * 4);
  float* TB  = (float*)carve(2ull * 9 * 1024 * 4);
  float* PB  = (float*)carve(2ull * 9 * 1024 * 4);
  float* LA  = (float*)carve(2ull * 21 * 1024 * 4);
  float* LB_ = (float*)carve(2ull * 21 * 1024 * 4);
  float* AFF = (float*)carve(2ull * 1024 * 1024 * 4);
  unsigned short* AFFB = (unsigned short*)carve(2ull * 1024 * 1024 * 2);
  unsigned short* XNB  = (unsigned short*)carve(2ull * 128 * 1024 * 2);
  unsigned short* LFB  = (unsigned short*)carve(2ull * 21 * 1024 * 2);
  float* POOL = (float*)carve(2ull * 2048 * 4);
  float* X4   = (float*)carve(2ull * 1024 * 1024 * 4);

  Ctx c{stream, XB, WB};
  const int B = 2;

  // ---- q path: probs -> qf0(s4) -> qf1(s4) -> qf2(s1) ----
  convOp(c, probs, qf[0].w, qf[0].b, qf[0].s, qf[0].bb, nullptr, P0,
         B, 9, 512, 512, 32, 4, 4, 1, 1);
  convOp(c, P0, qf[1].w, qf[1].b, qf[1].s, qf[1].bb, nullptr, P1,
         B, 32, 128, 128, 128, 4, 4, 1, 1);
  convOp(c, P1, qf[2].w, qf[2].b, qf[2].s, qf[2].bb, nullptr, Q,
         B, 128, 32, 32, 128, 3, 1, 1, 1);

  // ---- stem ----
  convOp(c, inputs, c1w, nullptr, s0, b0, nullptr, P0,
         B, 3, 512, 512, 64, 7, 2, 3, 1);
  {
    long tot = (long)B * 64 * 128 * 128;
    maxpool_kernel<<<(unsigned)((tot + 255) / 256), 256, 0, stream>>>(
        P0, P1, B * 64, 256, 256, 128, 128);
  }
  float* x = P1;
  int C = 64, H = 128, Wd = 128;
  auto runL = [&](BlockP* blks, int n, int mid, int stride) {
    for (int i = 0; i < n; ++i) {
      float* out = (x == P0) ? P1 : P0;
      int s = (i == 0) ? stride : 1;
      bneck(c, blks[i], x, out, T, S, B, C, H, Wd, mid, s);
      x = out;
      if (i == 0) { C = mid * 4; H = outDim(H, 3, s, 1); Wd = outDim(Wd, 3, s, 1); }
    }
  };
  runL(L1, 3, 64, 1);
  runL(L2, 4, 128, 2);
  runL(L3, 6, 256, 2);
  float* x4o = x;   // [2,1024,32,32]

  // ---- t4 head + upfeat -> X4 ----
  convOp(c, x4o, x4f.w, x4f.b, x4f.s, x4f.bb, nullptr, F,
         B, 1024, 32, 32, 128, 1, 1, 0, 1);
  {
    long tot = (long)B * 256 * 1024;
    concat2_pack_kernel<<<(unsigned)((tot + 255) / 256), 256, 0, stream>>>(
        F, Q, XB, B, 128, 128, 1024);
    convRaw(c, tc4[0].w, tc4[0].b, nullptr, nullptr, nullptr, T,
            B, 256, 32, 32, 128, 3, 1, 1, 1);
  }
  convOp(c, T, tc4[1].w, tc4[1].b, nullptr, nullptr, nullptr, TB,
         B, 128, 32, 32, 9, 1, 1, 0, 1);
  softmax9_kernel<<<(unsigned)((B * 1024 + 255) / 256), 256, 0, stream>>>(TB, PB, B, 1024);
  {
    long tot = (long)B * 1024 * 1024;
    upfeat_kernel<<<(unsigned)((tot + 255) / 256), 256, 0, stream>>>(
        x4o, PB, X4, B, 1024, 32, 32);
  }

  // ---- layer4 on X4 ----
  x = X4; C = 1024; H = 32; Wd = 32;
  runL(L4, 3, 512, 1);
  float* x5 = x;   // [2,2048,32,32]

  // ---- logits + t5 head + upfeat ----
  convOp(c, x5, clsw, nullptr, nullptr, nullptr, nullptr, LA,
         B, 2048, 32, 32, 21, 1, 1, 0, 0);
  convOp(c, x5, x5f.w, x5f.b, x5f.s, x5f.bb, nullptr, F,
         B, 2048, 32, 32, 128, 1, 1, 0, 1);
  {
    long tot = (long)B * 256 * 1024;
    concat2_pack_kernel<<<(unsigned)((tot + 255) / 256), 256, 0, stream>>>(
        F, Q, XB, B, 128, 128, 1024);
    convRaw(c, tc5[0].w, tc5[0].b, nullptr, nullptr, nullptr, T,
            B, 256, 32, 32, 128, 3, 1, 1, 1);
  }
  convOp(c, T, tc5[1].w, tc5[1].b, nullptr, nullptr, nullptr, TB,
         B, 128, 32, 32, 9, 1, 1, 0, 1);
  softmax9_kernel<<<(unsigned)((B * 1024 + 255) / 256), 256, 0, stream>>>(TB, PB, B, 1024);
  {
    long tot = (long)B * 21 * 1024;
    upfeat_kernel<<<(unsigned)((tot + 255) / 256), 256, 0, stream>>>(
        LA, PB, LB_, B, 21, 32, 32);
  }

  // ---- logits_min: global mean -> cls (written directly to d_out tail) ----
  avgpool_kernel<<<(unsigned)((B * 2048 + 255) / 256), 256, 0, stream>>>(x5, POOL, B * 2048, 1024);
  convOp(c, POOL, clsw, nullptr, nullptr, nullptr, nullptr,
         (float*)d_out + (long)B * 21 * 1024, B, 2048, 1, 1, 21, 1, 1, 0, 0);

  // ---- affinity refinement (pcm=2, it=2) ----
  l2norm_pack_kernel<<<(unsigned)((B * 1024 + 255) / 256), 256, 0, stream>>>(
      X4, XNB, B, 128, 1024);
  {
    // aff = (xf^T xf)^2 : A(i,c)=xn[c*1024+i], B(c,j)=xn[c*1024+j]
    GemmP g{XNB, XNB, AFF, 1, 1024, 1024, 1,
            128l * 1024, 128l * 1024, 1024l * 1024, 1024, 1024, 128, 1};
    gemm_wmma_kernel<<<dim3(16, 16, B), 128, 0, stream>>>(g);
  }
  colnorm_kernel<<<(unsigned)((B * 1024 + 255) / 256), 256, 0, stream>>>(AFF, B, 1024);
  launch_pack(AFF, AFFB, (long)B * 1024 * 1024, stream);
  long lfn = (long)B * 21 * 1024;
  launch_pack(LB_, LFB, lfn, stream);
  {
    GemmP g{LFB, AFFB, LA, 1024, 1, 1024, 1,
            21l * 1024, 1024l * 1024, 21l * 1024, 21, 1024, 1024, 0};
    gemm_wmma_kernel<<<dim3(16, 1, B), 128, 0, stream>>>(g);
  }
  launch_pack(LA, LFB, lfn, stream);
  {
    GemmP g{LFB, AFFB, (float*)d_out, 1024, 1, 1024, 1,
            21l * 1024, 1024l * 1024, 21l * 1024, 21, 1024, 1024, 0};
    gemm_wmma_kernel<<<dim3(16, 1, B), 128, 0, stream>>>(g);
  }
}
